// Net_25847113187867
// MI455X (gfx1250) — matrix operand order repaired and checked by reading, not compiled, not run on hardware
//
#include <hip/hip_runtime.h>
#include <hip/hip_bf16.h>
#include <math.h>

// ---------------------------------------------------------------------------
// Problem geometry
// ---------------------------------------------------------------------------
#define L_IN   16684              // input length
#define W_OUT  16384              // number of windows = L_IN - 300
#define NG     16688              // conv1 absolute positions (padded; valid p <= L_IN-5)
#define NQ2    16656              // conv2 absolute positions (padded, multiple of 16)
#define NQ3    16576              // conv3 absolute positions (padded, multiple of 16)

typedef __attribute__((ext_vector_type(16))) _Float16 v16h;
typedef __attribute__((ext_vector_type(8)))  _Float16 v8h;
typedef __attribute__((ext_vector_type(8)))  float    v8f;

// ---------------------------------------------------------------------------
// conv1 over the whole signal at stride 1 (dedup of all window taps).
// G[ch][p] = relu(b1[ch] + sum_{c,k} w1[ch,c,k] * x[c][p+k] * (c==0?sign0:1))
// ---------------------------------------------------------------------------
__global__ void k_conv1(const float* __restrict__ x, const float* __restrict__ w1,
                        const float* __restrict__ b1, _Float16* __restrict__ G,
                        float sign0)
{
    const int p  = blockIdx.x * blockDim.x + threadIdx.x;
    const int ch = blockIdx.y;
    if (p >= NG) return;
    float v = 0.0f;
    if (p <= L_IN - 5) {
        const float* w = w1 + ch * 10;
        float acc = b1[ch];
#pragma unroll
        for (int k = 0; k < 5; ++k) acc += w[k]     * (sign0 * x[p + k]);
#pragma unroll
        for (int k = 0; k < 5; ++k) acc += w[5 + k] * x[L_IN + p + k];
        v = fmaxf(acc, 0.0f);
    }
    G[(size_t)ch * NG + p] = (_Float16)v;
}

// ---------------------------------------------------------------------------
// Dilated im2col for conv2/conv3 as GEMM-B, packed K-contiguous per column:
// B[q][c*6+k] = S[c][q + dil*k]   (K = 384)
// blockDim = 384 (one thread per K index), gridDim = #positions
// ---------------------------------------------------------------------------
__global__ void k_im2col_conv(const _Float16* __restrict__ S, _Float16* __restrict__ B,
                              int srcLd, int dil)
{
    const int kidx = threadIdx.x;           // 0..383
    const int q    = blockIdx.x;
    const int c    = kidx / 6;
    const int k    = kidx - 6 * c;
    B[(size_t)q * 384 + kidx] = S[(size_t)c * srcLd + q + dil * k];
}

// ---------------------------------------------------------------------------
// fc1 input gather: feature f = c*7+m of window w is C3[c][w + 30*m]  (K = 448)
// ---------------------------------------------------------------------------
__global__ void k_im2col_fc1(const _Float16* __restrict__ C3, _Float16* __restrict__ B)
{
    const int f = threadIdx.x;              // 0..447
    const int w = blockIdx.x;
    const int c = f / 7;
    const int m = f - 7 * c;
    B[(size_t)w * 448 + f] = C3[(size_t)c * NQ3 + w + 30 * m];
}

// ---------------------------------------------------------------------------
// WMMA GEMM: D = relu(A(16x16 tile of [M][KPAD]) x B + bias).
// One wave per 16x16 D tile; K fully unrolled -> KPAD/32 v_wmma per wave.
// A fragment layout (16-bit A 16x32): lane = khalf*16 + m,
//   lanes 0-15 hold K 0..15, lanes 16-31 hold K 16..31 (matches B).
// B fragment: column n = lane&15, lanes 0-15 K 0..15, lanes 16-31 K 16..31,
//   loaded as one contiguous 32B read from the K-contiguous packed buffer.
// D layout: VGPR r -> row m0 + r + 8*(lane>>4), col n0 + (lane&15).
// MODE 0: f16 row-major [row*ldOut+col]   (conv feature maps)
// MODE 1: f16 col-packed [col*ldOut+row]  (feeds next GEMM's B directly)
// MODE 2: f32 col-major  [col*ldOut+row]  (fc2 output for projection)
// ---------------------------------------------------------------------------
template <int KPAD, int MODE>
__global__ __launch_bounds__(32)
void k_gemm(const _Float16* __restrict__ A, const _Float16* __restrict__ B,
            const float* __restrict__ bias,
            _Float16* __restrict__ dstH, float* __restrict__ dstF, int ldOut)
{
    const int lane  = threadIdx.x;
    const int n0    = blockIdx.x << 4;
    const int m0    = blockIdx.y << 4;
    const int l15   = lane & 15;
    const int khalf = lane >> 4;

    const _Float16* ap = A + (size_t)(m0 + l15) * KPAD + khalf * 16;
    const _Float16* bp = B + (size_t)(n0 + l15) * KPAD + khalf * 16;

    v8f acc = {};
#pragma unroll
    for (int kb = 0; kb < KPAD; kb += 32) {
        union { v16h v; struct { v8h lo, hi; } s; } ua;
        ua.s.lo = *reinterpret_cast<const v8h*>(ap + kb);
        ua.s.hi = *reinterpret_cast<const v8h*>(ap + kb + 8);
        const v16h bf = *reinterpret_cast<const v16h*>(bp + kb);
        acc = __builtin_amdgcn_wmma_f32_16x16x32_f16(false, ua.v, false, bf,
                                                     (short)0, acc, false, false);
    }

#pragma unroll
    for (int r = 0; r < 8; ++r) {
        const int row = m0 + r + (khalf << 3);
        const int col = n0 + l15;
        float v = fmaxf(acc[r] + bias[row], 0.0f);   // all four GEMMs have ReLU
        if (MODE == 0)      dstH[(size_t)row * ldOut + col] = (_Float16)v;
        else if (MODE == 1) dstH[(size_t)col * ldOut + row] = (_Float16)v;
        else                dstF[(size_t)col * ldOut + row] = v;
    }
}

// ---------------------------------------------------------------------------
// Weight f32 -> f16 repack (natural flatten already matches the GEMM K order)
// ---------------------------------------------------------------------------
__global__ void k_f32_to_f16(const float* __restrict__ s, _Float16* __restrict__ d, int n)
{
    const int i = blockIdx.x * blockDim.x + threadIdx.x;
    if (i < n) d[i] = (_Float16)s[i];
}

// ---------------------------------------------------------------------------
// Fold fc3 + RNN input projection (both linear, no ReLU between):
//   v[f] = sum_n wih[n]*fc3_w[n][f];  c0 = sum_n wih[n]*fc3_b[n] + bih + bhh
// ---------------------------------------------------------------------------
__global__ void k_proj_prep(const float* __restrict__ wih, const float* __restrict__ fc3w,
                            const float* __restrict__ fc3b, const float* __restrict__ bih,
                            const float* __restrict__ bhh, float* __restrict__ vc)
{
    const int f = threadIdx.x;          // blockDim = 96
    if (f < 80) {
        float a = 0.0f;
        for (int n = 0; n < 32; ++n) a += wih[n] * fc3w[n * 80 + f];
        vc[f] = a;
    } else if (f == 80) {
        float a = bih[0] + bhh[0];
        for (int n = 0; n < 32; ++n) a += wih[n] * fc3b[n];
        vc[80] = a;
    }
}

// xp[w] = c0 + v . H2[w]
__global__ void k_proj(const float* __restrict__ H2, const float* __restrict__ vc,
                       float* __restrict__ xp)
{
    const int w = blockIdx.x * blockDim.x + threadIdx.x;
    if (w >= W_OUT) return;
    const float* h = H2 + (size_t)w * 80;
    float a = vc[80];
#pragma unroll 8
    for (int j = 0; j < 80; ++j) a += vc[j] * h[j];
    xp[w] = a;
}

// ---------------------------------------------------------------------------
// Inherently sequential tanh recurrence; both passes run as concurrent waves.
// ---------------------------------------------------------------------------
__global__ void k_scan(const float* __restrict__ xpa, const float* __restrict__ xpb,
                       const float* __restrict__ whh,
                       float* __restrict__ ya, float* __restrict__ yb)
{
    const int t = threadIdx.x;          // blockDim = 64 -> two waves
    if (t != 0 && t != 32) return;
    const float wh  = whh[0];
    const float* xp = (t == 0) ? xpa : xpb;
    float*       y  = (t == 0) ? ya  : yb;
    float h = 0.0f;
    for (int i = 0; i < W_OUT; ++i) {
        h = tanhf(xp[i] + wh * h);
        y[i] = h;
    }
}

__global__ void k_combine(const float* __restrict__ ya, const float* __restrict__ yb,
                          float* __restrict__ out)
{
    const int i = blockIdx.x * blockDim.x + threadIdx.x;
    if (i < W_OUT) out[i] = 0.5f * (ya[i] - yb[i]);
}

// ---------------------------------------------------------------------------
// Host-side orchestration
// ---------------------------------------------------------------------------
extern "C" void kernel_launch(void* const* d_in, const int* in_sizes, int n_in,
                              void* d_out, int out_size, void* d_ws, size_t ws_size,
                              hipStream_t stream)
{
    (void)in_sizes; (void)n_in; (void)out_size; (void)ws_size;

    const float* x0  = (const float*)d_in[0];
    const float* w1  = (const float*)d_in[1];
    const float* b1  = (const float*)d_in[2];
    const float* w2  = (const float*)d_in[3];
    const float* b2  = (const float*)d_in[4];
    const float* w3  = (const float*)d_in[5];
    const float* b3  = (const float*)d_in[6];
    const float* fw1 = (const float*)d_in[7];
    const float* fb1 = (const float*)d_in[8];
    const float* fw2 = (const float*)d_in[9];
    const float* fb2 = (const float*)d_in[10];
    const float* fw3 = (const float*)d_in[11];
    const float* fb3 = (const float*)d_in[12];
    const float* wih = (const float*)d_in[13];
    const float* whh = (const float*)d_in[14];
    const float* bih = (const float*)d_in[15];
    const float* bhh = (const float*)d_in[16];
    float* out = (float*)d_out;

    // ---- workspace carve-up (256B aligned, stage-aliased; ~35.5 MB total) ----
    char*  base = (char*)d_ws;
    size_t off  = 0;
    auto take = [&](size_t bytes) -> char* {
        char* p = base + off;
        off += (bytes + 255) & ~(size_t)255;
        return p;
    };
    // bigA: conv2 im2col (384*NQ2) then fc1 im2col (448*W)
    _Float16* bigA = (_Float16*)take((size_t)448 * W_OUT * 2);
    // bigB: conv3 im2col (384*NQ3) then fc1 output packed (320*W)
    _Float16* bigB = (_Float16*)take((size_t)384 * NQ3 * 2);
    // gh2: conv1 features (64*NG f16) then fc2 output (W*80 f32)
    char*     gh2  = take((size_t)W_OUT * 80 * 4);
    // creg: conv2 features then conv3 features (64 * max(NQ2,NQ3) f16)
    _Float16* creg = (_Float16*)take((size_t)64 * NQ2 * 2);
    _Float16* w2p  = (_Float16*)take((size_t)64 * 384 * 2);
    _Float16* w3p  = (_Float16*)take((size_t)64 * 384 * 2);
    _Float16* fw1p = (_Float16*)take((size_t)320 * 448 * 2);
    _Float16* fw2p = (_Float16*)take((size_t)80 * 320 * 2);
    float*    vc   = (float*)take(128 * 4);
    float*    xpa  = (float*)take((size_t)W_OUT * 4);
    float*    xpb  = (float*)take((size_t)W_OUT * 4);
    float*    ya   = (float*)take((size_t)W_OUT * 4);
    float*    yb   = (float*)take((size_t)W_OUT * 4);

    // ---- weight repack (f32 -> f16; natural flatten == GEMM K-order) ----
    k_f32_to_f16<<<(24576 + 255) / 256, 256, 0, stream>>>(w2,  w2p,  24576);
    k_f32_to_f16<<<(24576 + 255) / 256, 256, 0, stream>>>(w3,  w3p,  24576);
    k_f32_to_f16<<<(143360 + 255) / 256, 256, 0, stream>>>(fw1, fw1p, 143360);
    k_f32_to_f16<<<(25600 + 255) / 256, 256, 0, stream>>>(fw2, fw2p, 25600);
    k_proj_prep<<<1, 96, 0, stream>>>(wih, fw3, fb3, bih, bhh, vc);

    // ---- two passes: sign0 = +1, then channel-0 negated ----
    for (int pass = 0; pass < 2; ++pass) {
        const float sign0 = (pass == 0) ? 1.0f : -1.0f;
        float* xp = (pass == 0) ? xpa : xpb;
        _Float16* G    = (_Float16*)gh2;
        float*    H2   = (float*)gh2;
        _Float16* B2   = bigA;
        _Float16* Bfc1 = bigA;
        _Float16* B3   = bigB;
        _Float16* H1p  = bigB;
        _Float16* C2   = creg;
        _Float16* C3   = creg;

        k_conv1<<<dim3((NG + 255) / 256, 64), 256, 0, stream>>>(x0, w1, b1, G, sign0);
        k_im2col_conv<<<NQ2, 384, 0, stream>>>(G, B2, NG, 5);
        k_gemm<384, 0><<<dim3(NQ2 / 16, 4), 32, 0, stream>>>(w2p, B2, b2, C2, nullptr, NQ2);
        k_im2col_conv<<<NQ3, 384, 0, stream>>>(C2, B3, NQ2, 15);
        k_gemm<384, 0><<<dim3(NQ3 / 16, 4), 32, 0, stream>>>(w3p, B3, b3, C3, nullptr, NQ3);
        k_im2col_fc1<<<W_OUT, 448, 0, stream>>>(C3, Bfc1);
        k_gemm<448, 1><<<dim3(W_OUT / 16, 20), 32, 0, stream>>>(fw1p, Bfc1, fb1, H1p, nullptr, 320);
        k_gemm<320, 2><<<dim3(W_OUT / 16, 5), 32, 0, stream>>>(fw2p, H1p, fb2, nullptr, H2, 80);
        k_proj<<<W_OUT / 256, 256, 0, stream>>>(H2, vc, xp);
    }

    k_scan<<<1, 64, 0, stream>>>(xpa, xpb, whh, ya, yb);
    k_combine<<<(W_OUT + 255) / 256, 256, 0, stream>>>(ya, yb, out);
}
